// GNN_23880018166151
// MI455X (gfx1250) — matrix operand order repaired and checked by reading, not compiled, not run on hardware
//
#include <hip/hip_runtime.h>
#include <hip/hip_bf16.h>
#include <math.h>

#define NNODES  50000
#define NEDGES  500000
#define HID     128
#define NTYPE   3
#define NREL    5
#define NHEADS  8
#define DKH     16
#define NLAYER  2
#define MAXT    240

typedef __attribute__((ext_vector_type(16))) _Float16 v16h;
typedef __attribute__((ext_vector_type(8)))  _Float16 v8h;
typedef __attribute__((ext_vector_type(8)))  float    v8f;
typedef __attribute__((ext_vector_type(8)))  unsigned v8u;

union U16 { v16h h; v8u u; };

__device__ __forceinline__ float gelu_exact(float x) {
    return 0.5f * x * (1.0f + erff(x * 0.70710678118654752f));
}

// order-preserving float<->uint encoding for atomicMax-based segment max
__device__ __forceinline__ unsigned fkey(float f) {
    unsigned b = __float_as_uint(f);
    return (b & 0x80000000u) ? ~b : (b | 0x80000000u);
}
__device__ __forceinline__ float funkey(unsigned k) {
    unsigned b = (k & 0x80000000u) ? (k ^ 0x80000000u) : ~k;
    return __uint_as_float(b);
}

// gfx1250 async bulk copy global->LDS (VGLOBAL: vdst=LDS offset, vaddr=i32
// offset, saddr=64-bit base). Tracked by ASYNCcnt.
__device__ __forceinline__ void async_load_b128(unsigned lds_off, unsigned gvoff,
                                                const void* base) {
    asm volatile("global_load_async_to_lds_b128 %0, %1, %2"
                 :: "v"(lds_off), "v"(gvoff), "s"(base) : "memory");
}
__device__ __forceinline__ void wait_async0() {
    asm volatile("s_wait_asynccnt 0x0" ::: "memory");
}

// ---------------------------------------------------------------------------
// Typed linear via WMMA: Out[row] = post( pre(X[row]) @ W[type[row]] + B[type[row]] )
// PRE:  0 = identity, 1 = exact gelu   |  POST: 0 none, 1 tanh, 2 skip-blend
// 256 threads / 64-row tile; wave w owns column tile w. All 12 B fragments
// (3 types x 4 K-steps) live in VGPRs, loaded as contiguous v16h from the
// pre-transposed f16 weights. Per-type row masking on packed u32 lanes chains
// all types into a single accumulator.
// ---------------------------------------------------------------------------
template<int PRE, int POST>
__global__ void __launch_bounds__(256)
typed_linear_wmma_kernel(const float*    __restrict__ X,
                         const int*      __restrict__ ntype,
                         const _Float16* __restrict__ WhT,   // [T,128(n),128(k)] f16 transposed
                         const float*    __restrict__ Bv,    // [T,128]
                         const float*    __restrict__ Xold,  // POST==2
                         const float*    __restrict__ skipv, // [T], POST==2
                         float* __restrict__ Out, int n)
{
    __shared__ __align__(16) float    Asf[64 * HID];     // raw f32 tile (async target)
    __shared__ __align__(16) _Float16 Ah[64][HID + 8];   // converted f16 tile
    __shared__ int stype[64];

    const int row0 = blockIdx.x * 64;
    const int tid  = threadIdx.x;            // 0..255
    const int wave = tid >> 5;               // column tile
    const int lane = tid & 31;

    // ---- async stage: the 64x128 f32 tile is contiguous (row stride == 128) ----
    {
        int r = tid >> 2;                    // row this thread's 128B chunk is in
        if (row0 + r < n) {
            unsigned lds0 = (unsigned)(size_t)(&Asf[0]) + (unsigned)(tid * 128);
            unsigned gv   = (unsigned)((size_t)row0 * 512u) + (unsigned)(tid * 128);
            #pragma unroll
            for (int j = 0; j < 8; ++j)
                async_load_b128(lds0 + j * 16, gv + j * 16, (const void*)X);
        }
        if (tid < 64) stype[tid] = (row0 + tid < n) ? ntype[row0 + tid] : 0;
    }
    wait_async0();
    __syncthreads();

    // ---- convert tile to f16 (optional pre-gelu) ----
    {
        int r  = tid >> 2;
        int c0 = (tid & 3) * 32;
        #pragma unroll
        for (int j = 0; j < 32; ++j) {
            float v = Asf[r * HID + c0 + j];
            if (PRE == 1) v = gelu_exact(v);
            Ah[r][c0 + j] = (_Float16)v;
        }
    }
    __syncthreads();

    const int half = lane >> 4;
    const int lrow = lane & 15;
    const int ncol = wave * 16 + lrow;

    // ---- preload all B fragments: contiguous 32B per lane from transposed W ----
    v16h bf[NTYPE][4];
    #pragma unroll
    for (int t = 0; t < NTYPE; ++t) {
        const _Float16* Wt = WhT + (size_t)t * HID * HID + (size_t)ncol * HID;
        #pragma unroll
        for (int k4 = 0; k4 < 4; ++k4) {
            const int kb = k4 * 32 + half * 16;   // lanes 0-15: K k0..k0+15; 16-31: +16
            bf[t][k4] = *(const v16h*)(Wt + kb);
        }
    }

    v8f acc[4];
    v8f zinit = {};
    #pragma unroll
    for (int rt = 0; rt < 4; ++rt) acc[rt] = zinit;

    #pragma unroll
    for (int rt = 0; rt < 4; ++rt) {
        const int arow = rt * 16 + lrow;          // A-frag row for this lane
        const int rty  = stype[rt * 16 + lrow];
        #pragma unroll
        for (int k4 = 0; k4 < 4; ++k4) {
            const int ka = k4 * 32 + half * 16;   // lanes 0-15: K 0-15; 16-31: 16-31
            v8h alo = *(const v8h*)&Ah[arow][ka];       // ds_load_b128
            v8h ahi = *(const v8h*)&Ah[arow][ka + 8];   // ds_load_b128
            U16 ua;
            #pragma unroll
            for (int j = 0; j < 8; ++j) { ua.h[j] = alo[j]; ua.h[8 + j] = ahi[j]; }
            #pragma unroll
            for (int t = 0; t < NTYPE; ++t) {
                const bool keep = (rty == t);     // per-lane-uniform row mask
                U16 um;
                #pragma unroll
                for (int j = 0; j < 8; ++j) um.u[j] = keep ? ua.u[j] : 0u;  // cndmask_b32
                acc[rt] = __builtin_amdgcn_wmma_f32_16x16x32_f16(
                            false, um.h, false, bf[t][k4], (short)0, acc[rt], false, false);
            }
        }
    }

    // ---- epilogue: bias (per row type) + activation / blend + store ----
    #pragma unroll
    for (int rt = 0; rt < 4; ++rt) {
        #pragma unroll
        for (int i = 0; i < 8; ++i) {
            int m   = rt * 16 + i + 8 * half;
            int row = row0 + m;
            if (row < n) {
                float v = acc[rt][i] + Bv[(size_t)stype[m] * HID + ncol];
                if (POST == 1) v = tanhf(v);
                if (POST == 2) {
                    float al = 1.0f / (1.0f + expf(-skipv[stype[m]]));
                    v = v * al + Xold[(size_t)row * HID + ncol] * (1.0f - al);
                }
                Out[(size_t)row * HID + ncol] = v;
            }
        }
    }
}

// one-time f32 -> f16 weight conversion with [k][n] -> [n][k] transpose per
// 128x128 matrix; out[m][n][k] = in[m][k][n]
__global__ void cvt_transpose_f16_kernel(const float* __restrict__ in,
                                         _Float16* __restrict__ out, int nMat)
{
    int gid = blockIdx.x * blockDim.x + threadIdx.x;
    if (gid >= nMat * HID * HID) return;
    int k  = gid & 127;
    int nn = (gid >> 7) & 127;
    int m  = gid >> 14;
    out[gid] = (_Float16)in[((size_t)m * HID + k) * HID + nn];
}

// rteP[time][col] = rte_b[col] + sum_d tab[time][d] * rte_w[d][col]
__global__ void rte_proj_kernel(const float* __restrict__ rw,  // [128][128]
                                const float* __restrict__ rb,  // [128]
                                float* __restrict__ rteP)      // [240][128]
{
    int gid = blockIdx.x * blockDim.x + threadIdx.x;
    if (gid >= MAXT * HID) return;
    int col = gid & (HID - 1);
    int tm  = gid >> 7;
    const float c     = -0.07195578575f;   // -ln(10000)/128
    const float inv_s =  0.08838834765f;   // 1/sqrt(128)
    float acc = rb[col];
    #pragma unroll 4
    for (int i = 0; i < HID / 2; ++i) {
        float div = expf(c * (float)(2 * i));
        float ph  = (float)tm * div;
        acc += sinf(ph) * inv_s * rw[(size_t)(2 * i)     * HID + col]
             + cosf(ph) * inv_s * rw[(size_t)(2 * i + 1) * HID + col];
    }
    rteP[gid] = acc;
}

// out[t][time][col] = sum_d rteP[time][d] * W[t][d][col]   (no bias)
__global__ void rte_kv_kernel(const float* __restrict__ rteP, // [240][128]
                              const float* __restrict__ W,    // [T][128][128]
                              float* __restrict__ out)        // [T][240][128]
{
    int gid = blockIdx.x * blockDim.x + threadIdx.x;
    if (gid >= NTYPE * MAXT * HID) return;
    int col = gid & (HID - 1);
    int tmp = gid >> 7;
    int tm  = tmp % MAXT;
    int t   = tmp / MAXT;
    const float* p = rteP + (size_t)tm * HID;
    const float* w = W + (size_t)t * HID * HID + col;
    float acc = 0.0f;
    #pragma unroll 4
    for (int dd = 0; dd < HID; ++dd) acc += p[dd] * w[(size_t)dd * HID];
    out[gid] = acc;
}

// Pass 1: logits + segment max (thread per (edge, head))
__global__ void att_logit_kernel(const float* __restrict__ kx,
                                 const float* __restrict__ qx,
                                 const float* __restrict__ rteK,  // [T][240][128]
                                 const int* __restrict__ src, const int* __restrict__ dst,
                                 const int* __restrict__ etype, const int* __restrict__ etime,
                                 const int* __restrict__ ntype,
                                 const float* __restrict__ rel_att, // [R][H][16][16]
                                 const float* __restrict__ rel_pri, // [R][H]
                                 float* __restrict__ logit,
                                 unsigned* __restrict__ amaxU)
{
    int gid = blockIdx.x * blockDim.x + threadIdx.x;
    if (gid >= NEDGES * NHEADS) return;
    int e = gid >> 3, h = gid & 7;
    int s = src[e], d = dst[e], r = etype[e], tm = etime[e];
    int st = ntype[s];
    const float4* kp = (const float4*)(kx   + (size_t)s * HID + h * DKH);
    const float4* rp = (const float4*)(rteK + ((size_t)st * MAXT + tm) * HID + h * DKH);
    const float4* qp = (const float4*)(qx   + (size_t)d * HID + h * DKH);
    float kv[16], qv[16];
    #pragma unroll
    for (int i = 0; i < 4; ++i) {
        float4 a = kp[i], b = rp[i], c = qp[i];
        kv[4*i+0] = a.x + b.x; kv[4*i+1] = a.y + b.y;
        kv[4*i+2] = a.z + b.z; kv[4*i+3] = a.w + b.w;
        qv[4*i+0] = c.x; qv[4*i+1] = c.y; qv[4*i+2] = c.z; qv[4*i+3] = c.w;
    }
    const float* A = rel_att + ((size_t)r * NHEADS + h) * (DKH * DKH);
    float acc = 0.0f;
    #pragma unroll 4
    for (int dd = 0; dd < 16; ++dd) {
        float kd = kv[dd];
        #pragma unroll
        for (int kk = 0; kk < 16; ++kk) acc += qv[kk] * A[dd * 16 + kk] * kd;
    }
    acc *= rel_pri[r * NHEADS + h] * 0.25f;   // 1/sqrt(DK)
    logit[gid] = acc;
    atomicMax(&amaxU[(size_t)d * NHEADS + h], fkey(acc));
}

// Pass 2: w = exp(logit - max), segment sum
__global__ void att_exp_kernel(const int* __restrict__ dst,
                               const unsigned* __restrict__ amaxU,
                               float* __restrict__ logit,
                               float* __restrict__ asum)
{
    int gid = blockIdx.x * blockDim.x + threadIdx.x;
    if (gid >= NEDGES * NHEADS) return;
    int e = gid >> 3, h = gid & 7;
    int d = dst[e];
    float m = funkey(amaxU[(size_t)d * NHEADS + h]);
    float w = expf(logit[gid] - m);
    logit[gid] = w;
    atomicAdd(&asum[(size_t)d * NHEADS + h], w);
}

// Pass 3: msg = (vx[src]+rteV) @ rel_msg[r,h]; aggr[dst] += msg * att
__global__ void aggregate_kernel(const float* __restrict__ vx,
                                 const float* __restrict__ rteV,
                                 const int* __restrict__ src, const int* __restrict__ dst,
                                 const int* __restrict__ etype, const int* __restrict__ etime,
                                 const int* __restrict__ ntype,
                                 const float* __restrict__ rel_msg, // [R][H][16][16]
                                 const float* __restrict__ logit_w,
                                 const float* __restrict__ asum,
                                 float* __restrict__ aggr)
{
    int gid = blockIdx.x * blockDim.x + threadIdx.x;
    if (gid >= NEDGES * NHEADS) return;
    int e = gid >> 3, h = gid & 7;
    int s = src[e], d = dst[e], r = etype[e], tm = etime[e];
    int st = ntype[s];
    float a = logit_w[gid] / (asum[(size_t)d * NHEADS + h] + 1e-16f);
    const float4* vp = (const float4*)(vx   + (size_t)s * HID + h * DKH);
    const float4* rp = (const float4*)(rteV + ((size_t)st * MAXT + tm) * HID + h * DKH);
    float vv[16];
    #pragma unroll
    for (int i = 0; i < 4; ++i) {
        float4 x = vp[i], y = rp[i];
        vv[4*i+0] = x.x + y.x; vv[4*i+1] = x.y + y.y;
        vv[4*i+2] = x.z + y.z; vv[4*i+3] = x.w + y.w;
    }
    const float* M = rel_msg + ((size_t)r * NHEADS + h) * (DKH * DKH);
    float* dp = aggr + (size_t)d * HID + h * DKH;
    #pragma unroll
    for (int kk = 0; kk < 16; ++kk) {
        float m = 0.0f;
        #pragma unroll
        for (int dd = 0; dd < 16; ++dd) m += vv[dd] * M[dd * 16 + kk];
        atomicAdd(&dp[kk], m * a);
    }
}

// Final L2 normalization: one wave per row
__global__ void l2norm_kernel(const float* __restrict__ x, float* __restrict__ out, int n)
{
    int row  = blockIdx.x * 4 + (threadIdx.x >> 5);
    int lane = threadIdx.x & 31;
    if (row >= n) return;
    const float4* p = (const float4*)(x + (size_t)row * HID);
    float4 v = p[lane];
    float s = v.x * v.x + v.y * v.y + v.z * v.z + v.w * v.w;
    #pragma unroll
    for (int off = 16; off > 0; off >>= 1) s += __shfl_xor(s, off, 32);
    float inv = 1.0f / sqrtf(s);
    float4 o; o.x = v.x * inv; o.y = v.y * inv; o.z = v.z * inv; o.w = v.w * inv;
    ((float4*)(out + (size_t)row * HID))[lane] = o;
}

// ---------------------------------------------------------------------------
extern "C" void kernel_launch(void* const* d_in, const int* in_sizes, int n_in,
                              void* d_out, int out_size, void* d_ws, size_t ws_size,
                              hipStream_t stream)
{
    (void)in_sizes; (void)n_in; (void)out_size; (void)ws_size;
    const float* node_feature = (const float*)d_in[0];
    const int*   node_type    = (const int*)  d_in[1];
    const int*   edge_index   = (const int*)  d_in[2];   // [2,E]
    const int*   edge_type    = (const int*)  d_in[3];
    const int*   edge_time    = (const int*)  d_in[4];
    const float* adapt_w = (const float*)d_in[5];
    const float* adapt_b = (const float*)d_in[6];
    const float* wk  = (const float*)d_in[7];
    const float* bk  = (const float*)d_in[8];
    const float* wq  = (const float*)d_in[9];
    const float* bq  = (const float*)d_in[10];
    const float* wv  = (const float*)d_in[11];
    const float* bv  = (const float*)d_in[12];
    const float* wa  = (const float*)d_in[13];
    const float* ba  = (const float*)d_in[14];
    const float* rel_att = (const float*)d_in[15];  // [L,R,H,16,16]
    const float* rel_msg = (const float*)d_in[16];
    const float* rel_pri = (const float*)d_in[17];  // [L,R,H]
    const float* skip    = (const float*)d_in[18];  // [L,T]
    const float* rte_w   = (const float*)d_in[19];  // [L,128,128]
    const float* rte_b   = (const float*)d_in[20];  // [L,128]

    const int* srcI = edge_index;
    const int* dstI = edge_index + NEDGES;

    // workspace layout (floats first, then transposed f16 weight copies)
    float* f = (float*)d_ws;
    float* xA   = f; f += (size_t)NNODES * HID;
    float* xB   = f; f += (size_t)NNODES * HID;
    float* kx   = f; f += (size_t)NNODES * HID;
    float* vx   = f; f += (size_t)NNODES * HID;
    float* qx   = f; f += (size_t)NNODES * HID;
    float* aggr = f; f += (size_t)NNODES * HID;
    float* rteP = f; f += (size_t)MAXT * HID;
    float* rteK = f; f += (size_t)NTYPE * MAXT * HID;
    float* rteV = f; f += (size_t)NTYPE * MAXT * HID;
    float* logit = f; f += (size_t)NEDGES * NHEADS;
    unsigned* amaxU = (unsigned*)f; f += (size_t)NNODES * NHEADS;
    float* asum = f; f += (size_t)NNODES * NHEADS;

    const size_t nAdaptW = (size_t)NTYPE * HID * HID;
    const size_t nLayerW = (size_t)NLAYER * NTYPE * HID * HID;
    _Float16* hb = (_Float16*)f;
    _Float16* adapt_wh = hb; hb += nAdaptW;
    _Float16* wk_h = hb; hb += nLayerW;
    _Float16* wq_h = hb; hb += nLayerW;
    _Float16* wv_h = hb; hb += nLayerW;
    _Float16* wa_h = hb; hb += nLayerW;

    const dim3 blkTL(256), grdTL((NNODES + 63) / 64);
    const dim3 blkE(256),  grdE((NEDGES * NHEADS + 255) / 256);

    // one-time weight conversion to transposed f16 ([T][n][k])
    cvt_transpose_f16_kernel<<<((int)nAdaptW + 255) / 256, 256, 0, stream>>>(adapt_w, adapt_wh, NTYPE);
    cvt_transpose_f16_kernel<<<((int)nLayerW + 255) / 256, 256, 0, stream>>>(wk, wk_h, NLAYER * NTYPE);
    cvt_transpose_f16_kernel<<<((int)nLayerW + 255) / 256, 256, 0, stream>>>(wq, wq_h, NLAYER * NTYPE);
    cvt_transpose_f16_kernel<<<((int)nLayerW + 255) / 256, 256, 0, stream>>>(wv, wv_h, NLAYER * NTYPE);
    cvt_transpose_f16_kernel<<<((int)nLayerW + 255) / 256, 256, 0, stream>>>(wa, wa_h, NLAYER * NTYPE);

    // adapt: x = tanh(typed_linear(node_feature))
    typed_linear_wmma_kernel<0, 1><<<grdTL, blkTL, 0, stream>>>(
        node_feature, node_type, adapt_wh, adapt_b, nullptr, nullptr, xA, NNODES);

    float* xcur = xA;
    float* xnext = xB;
    for (int l = 0; l < NLAYER; ++l) {
        const float* wk_l = wk + (size_t)l * NTYPE * HID * HID;
        const float* wv_l = wv + (size_t)l * NTYPE * HID * HID;
        const _Float16* wkh_l = wk_h + (size_t)l * NTYPE * HID * HID;
        const _Float16* wqh_l = wq_h + (size_t)l * NTYPE * HID * HID;
        const _Float16* wvh_l = wv_h + (size_t)l * NTYPE * HID * HID;
        const _Float16* wah_l = wa_h + (size_t)l * NTYPE * HID * HID;
        const float* bk_l = bk + (size_t)l * NTYPE * HID;
        const float* bq_l = bq + (size_t)l * NTYPE * HID;
        const float* bv_l = bv + (size_t)l * NTYPE * HID;
        const float* ba_l = ba + (size_t)l * NTYPE * HID;
        const float* ratt_l = rel_att + (size_t)l * NREL * NHEADS * DKH * DKH;
        const float* rmsg_l = rel_msg + (size_t)l * NREL * NHEADS * DKH * DKH;
        const float* rpri_l = rel_pri + (size_t)l * NREL * NHEADS;
        const float* skip_l = skip + (size_t)l * NTYPE;

        // temporal-encoding projections (tiny; f32 weights for accuracy)
        rte_proj_kernel<<<(MAXT * HID + 127) / 128, 128, 0, stream>>>(
            rte_w + (size_t)l * HID * HID, rte_b + (size_t)l * HID, rteP);
        rte_kv_kernel<<<(NTYPE * MAXT * HID + 255) / 256, 256, 0, stream>>>(rteP, wk_l, rteK);
        rte_kv_kernel<<<(NTYPE * MAXT * HID + 255) / 256, 256, 0, stream>>>(rteP, wv_l, rteV);

        // node-level typed projections (WMMA)
        typed_linear_wmma_kernel<0, 0><<<grdTL, blkTL, 0, stream>>>(
            xcur, node_type, wkh_l, bk_l, nullptr, nullptr, kx, NNODES);
        typed_linear_wmma_kernel<0, 0><<<grdTL, blkTL, 0, stream>>>(
            xcur, node_type, wvh_l, bv_l, nullptr, nullptr, vx, NNODES);
        typed_linear_wmma_kernel<0, 0><<<grdTL, blkTL, 0, stream>>>(
            xcur, node_type, wqh_l, bq_l, nullptr, nullptr, qx, NNODES);

        hipMemsetAsync(amaxU, 0, (size_t)NNODES * NHEADS * sizeof(unsigned), stream);
        hipMemsetAsync(asum,  0, (size_t)NNODES * NHEADS * sizeof(float), stream);
        hipMemsetAsync(aggr,  0, (size_t)NNODES * HID * sizeof(float), stream);

        att_logit_kernel<<<grdE, blkE, 0, stream>>>(
            kx, qx, rteK, srcI, dstI, edge_type, edge_time, node_type,
            ratt_l, rpri_l, logit, amaxU);
        att_exp_kernel<<<grdE, blkE, 0, stream>>>(dstI, amaxU, logit, asum);
        aggregate_kernel<<<grdE, blkE, 0, stream>>>(
            vx, rteV, srcI, dstI, edge_type, edge_time, node_type,
            rmsg_l, logit, asum, aggr);

        // trans = typed_linear(gelu(aggr)); x = trans*alpha + x*(1-alpha)
        typed_linear_wmma_kernel<1, 2><<<grdTL, blkTL, 0, stream>>>(
            aggr, node_type, wah_l, ba_l, xcur, skip_l, xnext, NNODES);

        float* tmp = xcur; xcur = xnext; xnext = tmp;
    }

    l2norm_kernel<<<(NNODES + 3) / 4, 128, 0, stream>>>(xcur, (float*)d_out, NNODES);
}
